// Pointnet2Backbone_57578331570260
// MI455X (gfx1250) — matrix operand order repaired and checked by reading, not compiled, not run on hardware
//
#include <hip/hip_runtime.h>
#include <hip/hip_bf16.h>

typedef __attribute__((ext_vector_type(16))) _Float16 v16h;
typedef __attribute__((ext_vector_type(8)))  float    v8f;

union Frag { v16h h; unsigned int u[8]; };

// ---------------------------------------------------------------------------
// Farthest point sampling: one block per batch, iterative argmax reduction.
// ---------------------------------------------------------------------------
__global__ void __launch_bounds__(1024)
fps_kernel(const float* __restrict__ src, int stride, float* __restrict__ dist,
           int* __restrict__ out, int N, int npoint) {
    int b = blockIdx.x;
    const float* P = src + (size_t)b * N * stride;
    float* D = dist + (size_t)b * N;
    int* O = out + (size_t)b * npoint;
    int t = threadIdx.x, nt = blockDim.x;

    for (int i = t; i < N; i += nt) D[i] = 1e10f;
    __syncthreads();

    __shared__ float s_v[32];
    __shared__ int   s_i[32];
    __shared__ int   s_sel;

    int sel = 0;
    for (int it = 0; it < npoint; ++it) {
        if (t == 0) O[it] = sel;
        float cx = P[(size_t)sel * stride + 0];
        float cy = P[(size_t)sel * stride + 1];
        float cz = P[(size_t)sel * stride + 2];
        float best = -1.0f; int bi = 0;
        for (int i = t; i < N; i += nt) {
            float dx = P[(size_t)i * stride + 0] - cx;
            float dy = P[(size_t)i * stride + 1] - cy;
            float dz = P[(size_t)i * stride + 2] - cz;
            float d = dx * dx + dy * dy + dz * dz;
            float dm = fminf(D[i], d);
            D[i] = dm;
            if (dm > best) { best = dm; bi = i; }
        }
        #pragma unroll
        for (int o = 16; o > 0; o >>= 1) {
            float ov = __shfl_down(best, o, 32);
            int   oi = __shfl_down(bi, o, 32);
            if (ov > best) { best = ov; bi = oi; }
        }
        int lane = t & 31, w = t >> 5;
        if (lane == 0) { s_v[w] = best; s_i[w] = bi; }
        __syncthreads();
        if (t == 0) {
            int nw = nt >> 5;
            float bv = s_v[0]; int bb = s_i[0];
            for (int k = 1; k < nw; ++k)
                if (s_v[k] > bv) { bv = s_v[k]; bb = s_i[k]; }
            s_sel = bb;
        }
        __syncthreads();
        sel = s_sel;
        __syncthreads();
    }
}

// ---------------------------------------------------------------------------
// Gather xyz coordinates for selected indices.
// ---------------------------------------------------------------------------
__global__ void gather_xyz_kernel(const float* __restrict__ src, int stride,
                                  const int* __restrict__ inds, float* __restrict__ out,
                                  int B, int np, int N) {
    int gid = blockIdx.x * blockDim.x + threadIdx.x;
    if (gid >= B * np) return;
    int b = gid / np;
    int i = inds[gid];
    const float* P = src + ((size_t)b * N + i) * stride;
    out[(size_t)gid * 3 + 0] = P[0];
    out[(size_t)gid * 3 + 1] = P[1];
    out[(size_t)gid * 3 + 2] = P[2];
}

// ---------------------------------------------------------------------------
// Ball query: first nsample indices (ascending) within radius; pad w/ first.
// ---------------------------------------------------------------------------
__global__ void ball_query_kernel(const float* __restrict__ src, int stride,
                                  const float* __restrict__ q, int* __restrict__ idx,
                                  int B, int N, int M, float r2, int ns) {
    int gid = blockIdx.x * blockDim.x + threadIdx.x;
    if (gid >= B * M) return;
    int b = gid / M;
    const float* P = src + (size_t)b * N * stride;
    const float* Q = q + (size_t)gid * 3;
    int* out = idx + (size_t)gid * ns;
    float qx = Q[0], qy = Q[1], qz = Q[2];
    int cnt = 0;
    for (int i = 0; i < N && cnt < ns; ++i) {
        float dx = P[(size_t)i * stride + 0] - qx;
        float dy = P[(size_t)i * stride + 1] - qy;
        float dz = P[(size_t)i * stride + 2] - qz;
        float d = dx * dx + dy * dy + dz * dz;
        if (d <= r2) out[cnt++] = i;
    }
    int f = (cnt > 0) ? out[0] : 0;
    for (int k = cnt; k < ns; ++k) out[k] = f;
}

// ---------------------------------------------------------------------------
// Group: build padded f16 feature rows [ (xyz-c)/r | feats | 0-pad ].
// ---------------------------------------------------------------------------
__global__ void group_kernel(const float* __restrict__ src, int sstride,
                             const float* __restrict__ feats, int fstride, int Cf,
                             const float* __restrict__ newxyz, const int* __restrict__ idx,
                             _Float16* __restrict__ out,
                             int total, int np, int ns, int Kp, float invr, int N) {
    int gid = blockIdx.x * blockDim.x + threadIdx.x;
    if (gid >= total) return;
    int p = (gid / ns) % np;
    int b = gid / (ns * np);
    int i = idx[gid];
    const float* P = src + ((size_t)b * N + i) * sstride;
    const float* C = newxyz + ((size_t)b * np + p) * 3;
    _Float16* o = out + (size_t)gid * Kp;
    o[0] = (_Float16)((P[0] - C[0]) * invr);
    o[1] = (_Float16)((P[1] - C[1]) * invr);
    o[2] = (_Float16)((P[2] - C[2]) * invr);
    int k = 3;
    if (feats) {
        const float* F = feats + ((size_t)b * N + i) * fstride;
        for (; k < 3 + Cf; ++k) o[k] = (_Float16)F[k - 3];
    }
    for (; k < Kp; ++k) o[k] = (_Float16)0.0f;
}

// ---------------------------------------------------------------------------
// Weight conversion: f32 (K, N) row-major -> f16 transposed (N, Kp),
// k-contiguous rows, zero-padded for k in [K, Kp).
// ---------------------------------------------------------------------------
__global__ void convert_w_kernel(const float* __restrict__ W, _Float16* __restrict__ Wh,
                                 int K, int Kp, int N) {
    int gid = blockIdx.x * blockDim.x + threadIdx.x;
    if (gid >= N * Kp) return;
    int k = gid % Kp;
    int n = gid / Kp;
    Wh[gid] = (k < K) ? (_Float16)W[(size_t)k * N + n] : (_Float16)0.0f;
}

// ---------------------------------------------------------------------------
// WMMA GEMM + scale/bias + ReLU:  Y = relu((X @ W) * g + b)
// X:  (M, Kp) f16 row-major (zero padded past K).
// Wh: (N, Kp) f16 (transposed, padded) - B fragments load contiguously.
// Block = 256 threads = 8 waves; each wave owns a distinct 16-row M tile and
// computes NSUB (compile-time) 16-col N subtiles per A fragment. Branch-free
// inner loop: one clause of b128 loads, one wait, NSUB back-to-back v_wmma.
// ---------------------------------------------------------------------------
template <int NSUB>
__global__ void __launch_bounds__(256)
gemm_relu_kernel(const _Float16* __restrict__ X, const _Float16* __restrict__ Wh,
                 const float* __restrict__ Gm, const float* __restrict__ Bt,
                 _Float16* __restrict__ Y, float* __restrict__ Yf,
                 int M, int Kp, int N, int n_origin) {
    int t = threadIdx.x;
    int lane = t & 31, wave = t >> 5;
    int m0 = blockIdx.x * 128 + wave * 16;
    int n0 = n_origin + blockIdx.y * 64;
    int nl = lane & 15;
    int kb   = (lane >> 4) * 8;   // A layout: upper half-wave holds K+8 block
    int koff = (lane >> 4) * 8;   // B layout: upper half-wave holds K+8 block

    const _Float16* Xrow = X + (size_t)(m0 + nl) * Kp;
    const _Float16* B0   = Wh + (size_t)(n0 + nl) * Kp + koff;

    v8f acc[NSUB];
    #pragma unroll
    for (int j = 0; j < NSUB; ++j) acc[j] = (v8f){};

    for (int k0 = 0; k0 < Kp; k0 += 32) {
        Frag a;
        #pragma unroll
        for (int v = 0; v < 4; ++v) {
            int ka = k0 + kb + 2 * v;
            a.u[v]     = *(const unsigned int*)(Xrow + ka);
            a.u[v + 4] = *(const unsigned int*)(Xrow + ka + 16);
        }
        Frag bf[NSUB];
        #pragma unroll
        for (int j = 0; j < NSUB; ++j) {
            const _Float16* Bp = B0 + (size_t)j * 16 * Kp + k0;
            #pragma unroll
            for (int v = 0; v < 4; ++v) {
                bf[j].u[v]     = *(const unsigned int*)(Bp + 2 * v);
                bf[j].u[v + 4] = *(const unsigned int*)(Bp + 2 * v + 16);
            }
        }
        __builtin_prefetch(Xrow + k0 + 32, 0, 0);
        #pragma unroll
        for (int j = 0; j < NSUB; ++j) {
            acc[j] = __builtin_amdgcn_wmma_f32_16x16x32_f16(
                /*neg_a=*/false, a.h, /*neg_b=*/false, bf[j].h,
                /*c_mod=*/(short)0, acc[j], /*reuse_a=*/false, /*reuse_b=*/false);
        }
    }

    int mb = m0 + ((lane >> 4) << 3);
    #pragma unroll
    for (int j = 0; j < NSUB; ++j) {
        int n = n0 + 16 * j + nl;
        float gv = Gm[n], bv = Bt[n];
        #pragma unroll
        for (int r = 0; r < 8; ++r) {
            float y = acc[j][r] * gv + bv;
            y = y > 0.0f ? y : 0.0f;
            size_t o = (size_t)(mb + r) * N + n;
            Y[o] = (_Float16)y;
            if (Yf) Yf[o] = y;
        }
    }
}

// ---------------------------------------------------------------------------
// Max-pool over nsample dimension: f16 (pts*ns, C) -> f32 (pts, C).
// ---------------------------------------------------------------------------
__global__ void maxpool_kernel(const _Float16* __restrict__ Y, float* __restrict__ out,
                               int pts, int ns, int C) {
    int gid = blockIdx.x * blockDim.x + threadIdx.x;
    if (gid >= pts * C) return;
    int c = gid % C, p = gid / C;
    const _Float16* src = Y + (size_t)p * ns * C + c;
    float m = -3.0e38f;
    for (int s = 0; s < ns; ++s) {
        float v = (float)src[(size_t)s * C];
        m = v > m ? v : m;
    }
    out[gid] = m;
}

// ---------------------------------------------------------------------------
// 3-NN with inverse-distance weights.
// ---------------------------------------------------------------------------
__global__ void knn3_kernel(const float* __restrict__ unk, const float* __restrict__ kn,
                            int* __restrict__ idx3, float* __restrict__ w3,
                            int B, int nu, int nk) {
    int gid = blockIdx.x * blockDim.x + threadIdx.x;
    if (gid >= B * nu) return;
    int b = gid / nu;
    const float* U = unk + (size_t)gid * 3;
    const float* Kx = kn + (size_t)b * nk * 3;
    float ux = U[0], uy = U[1], uz = U[2];
    float d0 = 3e38f, d1 = 3e38f, d2 = 3e38f;
    int i0 = 0, i1 = 0, i2 = 0;
    for (int i = 0; i < nk; ++i) {
        float dx = Kx[(size_t)i * 3 + 0] - ux;
        float dy = Kx[(size_t)i * 3 + 1] - uy;
        float dz = Kx[(size_t)i * 3 + 2] - uz;
        float d = dx * dx + dy * dy + dz * dz;
        if (d < d0)      { d2 = d1; i2 = i1; d1 = d0; i1 = i0; d0 = d; i0 = i; }
        else if (d < d1) { d2 = d1; i2 = i1; d1 = d;  i1 = i; }
        else if (d < d2) { d2 = d;  i2 = i; }
    }
    float w0 = 1.0f / (sqrtf(fmaxf(d0, 0.0f)) + 1e-8f);
    float w1 = 1.0f / (sqrtf(fmaxf(d1, 0.0f)) + 1e-8f);
    float w2 = 1.0f / (sqrtf(fmaxf(d2, 0.0f)) + 1e-8f);
    float s = w0 + w1 + w2;
    idx3[(size_t)gid * 3 + 0] = i0;
    idx3[(size_t)gid * 3 + 1] = i1;
    idx3[(size_t)gid * 3 + 2] = i2;
    w3[(size_t)gid * 3 + 0] = w0 / s;
    w3[(size_t)gid * 3 + 1] = w1 / s;
    w3[(size_t)gid * 3 + 2] = w2 / s;
}

// ---------------------------------------------------------------------------
// FP concat: X = [interp(kn_feats), unk_feats] as f16 (B*nu, C+Cu).
// ---------------------------------------------------------------------------
__global__ void fp_concat_kernel(const float* __restrict__ knf, const float* __restrict__ unkf,
                                 const int* __restrict__ idx3, const float* __restrict__ w3,
                                 _Float16* __restrict__ X, int B, int nu, int nk, int C, int Cu) {
    int tot = B * nu * (C + Cu);
    int gid = blockIdx.x * blockDim.x + threadIdx.x;
    if (gid >= tot) return;
    int ch = gid % (C + Cu);
    int p = gid / (C + Cu);
    int b = p / nu;
    _Float16* o = X + (size_t)p * (C + Cu) + ch;
    if (ch < C) {
        const int* I = idx3 + (size_t)p * 3;
        const float* Wt = w3 + (size_t)p * 3;
        const float* KF = knf + (size_t)b * nk * C;
        float v = Wt[0] * KF[(size_t)I[0] * C + ch]
                + Wt[1] * KF[(size_t)I[1] * C + ch]
                + Wt[2] * KF[(size_t)I[2] * C + ch];
        *o = (_Float16)v;
    } else {
        *o = (_Float16)unkf[(size_t)p * Cu + (ch - C)];
    }
}

// ---------------------------------------------------------------------------
// Tail: append xyz2 and fp2_inds (= inds1[:, :1024]) to d_out.
// ---------------------------------------------------------------------------
__global__ void tail_kernel(const float* __restrict__ xyz2, const int* __restrict__ inds1,
                            float* __restrict__ out, int nxyz, int ninds) {
    int gid = blockIdx.x * blockDim.x + threadIdx.x;
    if (gid < nxyz) {
        out[gid] = xyz2[gid];
    } else if (gid < nxyz + ninds) {
        int j = gid - nxyz;
        int b = j / 1024, i = j % 1024;
        out[gid] = __int_as_float(inds1[b * 2048 + i]);
    }
}

// ---------------------------------------------------------------------------
// Host-side launch
// ---------------------------------------------------------------------------
static inline void launch_gemm(const _Float16* X, const float* W, const float* g,
                               const float* b, _Float16* Wh, _Float16* Y, float* Yf,
                               int M, int K, int Kp, int N, hipStream_t s) {
    int tot = N * Kp;
    convert_w_kernel<<<(tot + 255) / 256, 256, 0, s>>>(W, Wh, K, Kp, N);
    int nfull = N / 64;
    if (nfull > 0) {
        dim3 grid(M / 128, nfull);
        gemm_relu_kernel<4><<<grid, dim3(256), 0, s>>>(X, Wh, g, b, Y, Yf, M, Kp, N, 0);
    }
    int rem = (N % 64) / 16;
    int nb = nfull * 64;
    dim3 gridT(M / 128, 1);
    if (rem == 1) gemm_relu_kernel<1><<<gridT, dim3(256), 0, s>>>(X, Wh, g, b, Y, Yf, M, Kp, N, nb);
    else if (rem == 2) gemm_relu_kernel<2><<<gridT, dim3(256), 0, s>>>(X, Wh, g, b, Y, Yf, M, Kp, N, nb);
    else if (rem == 3) gemm_relu_kernel<3><<<gridT, dim3(256), 0, s>>>(X, Wh, g, b, Y, Yf, M, Kp, N, nb);
}

extern "C" void kernel_launch(void* const* d_in, const int* in_sizes, int n_in,
                              void* d_out, int out_size, void* d_ws, size_t ws_size,
                              hipStream_t stream) {
    const int B = 2, N0 = 20000;
    const float* pc = (const float*)d_in[0];

    // Flatten order per setup_inputs: pointcloud, then each params tuple as
    // (W0,g0,b0, W1,g1,b1, [W2,g2,b2], key).
    const float* saW[4][3]; const float* saG[4][3]; const float* saB[4][3];
    int base = 1;
    for (int l = 0; l < 4; ++l) {
        for (int j = 0; j < 3; ++j) {
            saW[l][j] = (const float*)d_in[base + 3 * j + 0];
            saG[l][j] = (const float*)d_in[base + 3 * j + 1];
            saB[l][j] = (const float*)d_in[base + 3 * j + 2];
        }
        base += 10;  // 9 arrays + key
    }
    const float* fpW[2][2]; const float* fpG[2][2]; const float* fpB[2][2];
    for (int l = 0; l < 2; ++l) {
        for (int j = 0; j < 2; ++j) {
            fpW[l][j] = (const float*)d_in[base + 3 * j + 0];
            fpG[l][j] = (const float*)d_in[base + 3 * j + 1];
            fpB[l][j] = (const float*)d_in[base + 3 * j + 2];
        }
        base += 7;  // 6 arrays + key
    }

    // ---- workspace layout --------------------------------------------------
    char* ws = (char*)d_ws;
    size_t off = 0;
    auto take = [&](size_t bytes) -> void* {
        void* p = ws + off;
        off += (bytes + 255) & ~(size_t)255;
        return p;
    };
    float* dist  = (float*)take((size_t)B * N0 * 4);
    int* inds1   = (int*)take((size_t)B * 2048 * 4);
    int* indsT   = (int*)take((size_t)B * 2048 * 4);
    float* xyz1  = (float*)take((size_t)B * 2048 * 3 * 4);
    float* xyz2  = (float*)take((size_t)B * 1024 * 3 * 4);
    float* xyz3  = (float*)take((size_t)B * 512 * 3 * 4);
    float* xyz4  = (float*)take((size_t)B * 256 * 3 * 4);
    int* idxg    = (int*)take((size_t)B * 2048 * 64 * 4);
    float* f1    = (float*)take((size_t)B * 2048 * 128 * 4);
    float* f2    = (float*)take((size_t)B * 1024 * 256 * 4);
    float* f3    = (float*)take((size_t)B * 512 * 256 * 4);
    float* f4    = (float*)take((size_t)B * 256 * 256 * 4);
    float* f3u   = (float*)take((size_t)B * 512 * 256 * 4);
    int* knni    = (int*)take((size_t)B * 1024 * 3 * 4);
    float* knnw  = (float*)take((size_t)B * 1024 * 3 * 4);
    _Float16* Whs  = (_Float16*)take((size_t)512 * 512 * 2);   // transposed weights
    _Float16* bufA = (_Float16*)take((size_t)262144 * 64 * 2);
    _Float16* bufB = (_Float16*)take((size_t)262144 * 128 * 2);

    auto cdiv = [](int a, int b) { return (a + b - 1) / b; };

    // =========================== SA1 ========================================
    {
        const int np = 2048, ns = 64, M = B * np * ns, Kp = 32;
        fps_kernel<<<B, 1024, 0, stream>>>(pc, 6, dist, inds1, N0, np);
        gather_xyz_kernel<<<cdiv(B * np, 256), 256, 0, stream>>>(pc, 6, inds1, xyz1, B, np, N0);
        ball_query_kernel<<<cdiv(B * np, 256), 256, 0, stream>>>(pc, 6, xyz1, idxg, B, N0, np, 0.04f, ns);
        group_kernel<<<cdiv(M, 256), 256, 0, stream>>>(pc, 6, pc + 3, 6, 3, xyz1, idxg, bufA,
                                                       M, np, ns, Kp, 5.0f, N0);
        launch_gemm(bufA, saW[0][0], saG[0][0], saB[0][0], Whs, bufB, nullptr, M, 6,   Kp,  64, stream);
        launch_gemm(bufB, saW[0][1], saG[0][1], saB[0][1], Whs, bufA, nullptr, M, 64,  64,  64, stream);
        launch_gemm(bufA, saW[0][2], saG[0][2], saB[0][2], Whs, bufB, nullptr, M, 64,  64, 128, stream);
        maxpool_kernel<<<cdiv(B * np * 128, 256), 256, 0, stream>>>(bufB, f1, B * np, ns, 128);
    }
    // =========================== SA2 ========================================
    {
        const int np = 1024, ns = 32, M = B * np * ns, Kp = 160;
        fps_kernel<<<B, 1024, 0, stream>>>(xyz1, 3, dist, indsT, 2048, np);
        gather_xyz_kernel<<<cdiv(B * np, 256), 256, 0, stream>>>(xyz1, 3, indsT, xyz2, B, np, 2048);
        ball_query_kernel<<<cdiv(B * np, 256), 256, 0, stream>>>(xyz1, 3, xyz2, idxg, B, 2048, np, 0.16f, ns);
        group_kernel<<<cdiv(M, 256), 256, 0, stream>>>(xyz1, 3, f1, 128, 128, xyz2, idxg, bufA,
                                                       M, np, ns, Kp, 2.5f, 2048);
        launch_gemm(bufA, saW[1][0], saG[1][0], saB[1][0], Whs, bufB, nullptr, M, 131, Kp, 128, stream);
        launch_gemm(bufB, saW[1][1], saG[1][1], saB[1][1], Whs, bufA, nullptr, M, 128, 128, 128, stream);
        launch_gemm(bufA, saW[1][2], saG[1][2], saB[1][2], Whs, bufB, nullptr, M, 128, 128, 256, stream);
        maxpool_kernel<<<cdiv(B * np * 256, 256), 256, 0, stream>>>(bufB, f2, B * np, ns, 256);
    }
    // =========================== SA3 ========================================
    {
        const int np = 512, ns = 16, M = B * np * ns, Kp = 288;
        fps_kernel<<<B, 1024, 0, stream>>>(xyz2, 3, dist, indsT, 1024, np);
        gather_xyz_kernel<<<cdiv(B * np, 256), 256, 0, stream>>>(xyz2, 3, indsT, xyz3, B, np, 1024);
        ball_query_kernel<<<cdiv(B * np, 256), 256, 0, stream>>>(xyz2, 3, xyz3, idxg, B, 1024, np, 0.64f, ns);
        group_kernel<<<cdiv(M, 256), 256, 0, stream>>>(xyz2, 3, f2, 256, 256, xyz3, idxg, bufA,
                                                       M, np, ns, Kp, 1.25f, 1024);
        launch_gemm(bufA, saW[2][0], saG[2][0], saB[2][0], Whs, bufB, nullptr, M, 259, Kp, 128, stream);
        launch_gemm(bufB, saW[2][1], saG[2][1], saB[2][1], Whs, bufA, nullptr, M, 128, 128, 128, stream);
        launch_gemm(bufA, saW[2][2], saG[2][2], saB[2][2], Whs, bufB, nullptr, M, 128, 128, 256, stream);
        maxpool_kernel<<<cdiv(B * np * 256, 256), 256, 0, stream>>>(bufB, f3, B * np, ns, 256);
    }
    // =========================== SA4 ========================================
    {
        const int np = 256, ns = 16, M = B * np * ns, Kp = 288;
        fps_kernel<<<B, 1024, 0, stream>>>(xyz3, 3, dist, indsT, 512, np);
        gather_xyz_kernel<<<cdiv(B * np, 256), 256, 0, stream>>>(xyz3, 3, indsT, xyz4, B, np, 512);
        ball_query_kernel<<<cdiv(B * np, 256), 256, 0, stream>>>(xyz3, 3, xyz4, idxg, B, 512, np, 1.44f, ns);
        group_kernel<<<cdiv(M, 256), 256, 0, stream>>>(xyz3, 3, f3, 256, 256, xyz4, idxg, bufA,
                                                       M, np, ns, Kp, 0.8333333f, 512);
        launch_gemm(bufA, saW[3][0], saG[3][0], saB[3][0], Whs, bufB, nullptr, M, 259, Kp, 128, stream);
        launch_gemm(bufB, saW[3][1], saG[3][1], saB[3][1], Whs, bufA, nullptr, M, 128, 128, 128, stream);
        launch_gemm(bufA, saW[3][2], saG[3][2], saB[3][2], Whs, bufB, nullptr, M, 128, 128, 256, stream);
        maxpool_kernel<<<cdiv(B * np * 256, 256), 256, 0, stream>>>(bufB, f4, B * np, ns, 256);
    }
    // =========================== FP1: xyz4->xyz3 ============================
    {
        const int nu = 512, nk = 256, M = B * nu;
        knn3_kernel<<<cdiv(B * nu, 256), 256, 0, stream>>>(xyz3, xyz4, knni, knnw, B, nu, nk);
        fp_concat_kernel<<<cdiv(B * nu * 512, 256), 256, 0, stream>>>(f4, f3, knni, knnw, bufA,
                                                                      B, nu, nk, 256, 256);
        launch_gemm(bufA, fpW[0][0], fpG[0][0], fpB[0][0], Whs, bufB, nullptr, M, 512, 512, 256, stream);
        launch_gemm(bufB, fpW[0][1], fpG[0][1], fpB[0][1], Whs, bufA, f3u,     M, 256, 256, 256, stream);
    }
    // =========================== FP2: xyz3->xyz2 ============================
    {
        const int nu = 1024, nk = 512, M = B * nu;
        knn3_kernel<<<cdiv(B * nu, 256), 256, 0, stream>>>(xyz2, xyz3, knni, knnw, B, nu, nk);
        fp_concat_kernel<<<cdiv(B * nu * 512, 256), 256, 0, stream>>>(f3u, f2, knni, knnw, bufA,
                                                                      B, nu, nk, 256, 256);
        launch_gemm(bufA, fpW[1][0], fpG[1][0], fpB[1][0], Whs, bufB, nullptr,       M, 512, 512, 256, stream);
        launch_gemm(bufB, fpW[1][1], fpG[1][1], fpB[1][1], Whs, bufA, (float*)d_out, M, 256, 256, 288, stream);
    }
    // =========================== Tail =======================================
    {
        int nxyz = B * 1024 * 3, ninds = B * 1024;
        float* tail = (float*)d_out + (size_t)B * 1024 * 288;
        tail_kernel<<<cdiv(nxyz + ninds, 256), 256, 0, stream>>>(xyz2, inds1, tail, nxyz, ninds);
    }
    (void)in_sizes; (void)n_in; (void)out_size; (void)ws_size;
}
